// ClassificationNet_22110491640138
// MI455X (gfx1250) — matrix operand order repaired and checked by reading, not compiled, not run on hardware
//
#include <hip/hip_runtime.h>
#include <hip/hip_bf16.h>

typedef _Float16 v16h __attribute__((ext_vector_type(16)));
typedef _Float16 v8h  __attribute__((ext_vector_type(8)));
typedef float    v8f  __attribute__((ext_vector_type(8)));

#define WMMA_F16(A, B, C) \
  __builtin_amdgcn_wmma_f32_16x16x32_f16(false, (A), false, (B), (short)0, (C), false, false)

// ---------------------------------------------------------------------------
// Fold fb (9,9,21) and harmonic_kernel (63,32) into W2 (243 x 32) f32:
// W2[(p*9+q)*3+c][o] = sum_t fb[p,q,t] * hk[c*21+t][o]
// ---------------------------------------------------------------------------
__global__ void compute_w2_kernel(const float* __restrict__ fb,
                                  const float* __restrict__ hk,
                                  float* __restrict__ w2) {
  int idx = blockIdx.x * blockDim.x + threadIdx.x;
  if (idx >= 243 * 32) return;
  int o = idx & 31;
  int f = idx >> 5;
  int c = f % 3;
  int pq = f / 3;
  float acc = 0.f;
#pragma unroll
  for (int t = 0; t < 21; ++t)
    acc += fb[pq * 21 + t] * hk[(c * 21 + t) * 32 + o];
  w2[f * 32 + o] = acc;
}

// ---------------------------------------------------------------------------
// Pack a row-major f32 weight matrix W[K x N] into fragment-ready f16 layout:
// out[(nt*ksteps + s)*32 + lane] holds 8 dwords = K-pairs for that lane.
// B-fragment layout (16-bit B, 32xK16): lane L -> column n = nt*16 + (L&15),
// kbase = (L&16)?8:0, k = s*32+kb+2(i&3)+((i&4)?16:0), dword i = { (k, n), (k+1, n) }.
// ---------------------------------------------------------------------------
__global__ void pack_b_kernel(const float* __restrict__ W, _Float16* __restrict__ out,
                              int Kreal, int Nreal, int ksteps, int ntiles) {
  int idx = blockIdx.x * blockDim.x + threadIdx.x;  // one dword (pair) per thread
  int total = ntiles * ksteps * 256;
  if (idx >= total) return;
  int i = idx & 7;
  int lane = (idx >> 3) & 31;
  int rest = idx >> 8;
  int s = rest % ksteps;
  int nt = rest / ksteps;
  int n = nt * 16 + (lane & 15);
  int kb = (lane & 16) ? 8 : 0;
  int k0 = s * 32 + kb + 2 * (i & 3) + ((i & 4) ? 16 : 0);
  _Float16 v0 = (k0 < Kreal && n < Nreal) ? (_Float16)W[k0 * Nreal + n] : (_Float16)0.f;
  _Float16 v1 = (k0 + 1 < Kreal && n < Nreal) ? (_Float16)W[(k0 + 1) * Nreal + n] : (_Float16)0.f;
  out[idx * 2]     = v0;
  out[idx * 2 + 1] = v1;
}

// ---------------------------------------------------------------------------
// Fused deformable layer: one wave handles 16 output pixels (m0..m0+15).
// Stage 1: im2col(x, 9x9 VALID) -> LDS A tile (16x256 f16); WMMA GEMM with
//          packed offset weights -> 16x176 offsets (+offset_bias) in LDS.
// Stage 2: 81-tap bilinear gather from x -> pix tile (16x243 f16) in LDS.
// Stage 3: WMMA GEMM pix @ W2 (243x32) + bias, ReLU -> h1 f16 [B*88*88][32].
// ---------------------------------------------------------------------------
__global__ __launch_bounds__(32) void deform_kernel(
    const float* __restrict__ x, const _Float16* __restrict__ bpo,
    const _Float16* __restrict__ bpw2, const float* __restrict__ obias,
    const float* __restrict__ hbias, _Float16* __restrict__ h1) {
  __shared__ __align__(16) _Float16 At[16][256];
  __shared__ float Ot[16][176];

  const int lane = threadIdx.x;
  const int m0 = blockIdx.x * 16;
  const int p = lane & 15;
  const int mp = m0 + p;
  const int pj = mp % 88;
  const int pi = (mp / 88) % 88;
  const int pb = mp / (88 * 88);

  // zero the K padding tail (243..255) of every row
  for (int idx = lane; idx < 16 * 13; idx += 32)
    At[idx / 13][243 + idx % 13] = (_Float16)0.f;

  // im2col of x (VALID, so always in-bounds): f = (dy*9+dx)*3 + c
  {
    int pos0 = (lane < 16) ? 0 : 41;
    int pos1 = (lane < 16) ? 41 : 81;
    for (int pos = pos0; pos < pos1; ++pos) {
      int dy = pos / 9, dx = pos % 9;
      const float* xp = x + (((pb * 96) + pi + dy) * 96 + (pj + dx)) * 3;
      At[p][pos * 3 + 0] = (_Float16)xp[0];
      At[p][pos * 3 + 1] = (_Float16)xp[1];
      At[p][pos * 3 + 2] = (_Float16)xp[2];
    }
  }
  __syncthreads();

  const int kba  = (lane & 16) ? 8 : 0;   // A-fragment K base
  const int nloc = lane & 15;             // C/D column
  const int hi8  = (lane & 16) ? 8 : 0;   // C/D row offset

  // Stage 1 GEMM: 11 N-tiles x 8 K-steps = 88 WMMAs
  for (int nt = 0; nt < 11; ++nt) {
    v8f acc = {0.f, 0.f, 0.f, 0.f, 0.f, 0.f, 0.f, 0.f};
#pragma unroll
    for (int s = 0; s < 8; ++s) {
      v8h lo = *(const v8h*)&At[p][s * 32 + kba];
      v8h hi = *(const v8h*)&At[p][s * 32 + 16 + kba];
      v16h a = __builtin_shufflevector(lo, hi, 0, 1, 2, 3, 4, 5, 6, 7,
                                       8, 9, 10, 11, 12, 13, 14, 15);
      v16h bf = *(const v16h*)(bpo + ((nt * 8 + s) * 32 + lane) * 16);
      acc = WMMA_F16(a, bf, acc);
    }
    int n = nt * 16 + nloc;
#pragma unroll
    for (int r = 0; r < 8; ++r) {
      float v = (n < 162) ? (acc[r] + obias[n]) : 0.f;
      Ot[r + hi8][n] = v;
    }
  }
  __syncthreads();

  // Stage 2: bilinear sampling -> pix tile (overwrites At[..][0..242])
  for (int idx = lane; idx < 16 * 81; idx += 32) {
    int pp = idx / 81, t = idx - pp * 81;
    int m = m0 + pp;
    int j = m % 88;
    int i = (m / 88) % 88;
    int b = m / (88 * 88);
    float yof = Ot[pp][2 * t];
    float xof = Ot[pp][2 * t + 1];
    float y  = fminf(fmaxf((float)(i + t / 9) + yof, 0.f), 95.f);
    float xx = fminf(fmaxf((float)(j + t % 9) + xof, 0.f), 95.f);
    int y0 = (int)floorf(y);
    int x0 = (int)floorf(xx);
    int y1 = min(y0 + 1, 95);
    int x1 = min(x0 + 1, 95);
    y0 = min(max(y0, 0), 95);
    x0 = min(max(x0, 0), 95);
    float y0f = (float)y0, y1f = (float)y1, x0f = (float)x0, x1f = (float)x1;
    float w0 = (y1f - y) * (x1f - xx);
    float w1 = (y1f - y) * (xx - x0f);
    float w2 = (y - y0f) * (x1f - xx);
    float w3 = (y - y0f) * (xx - x0f);
    const float* b00 = x + ((b * 96 + y0) * 96 + x0) * 3;
    const float* b01 = x + ((b * 96 + y0) * 96 + x1) * 3;
    const float* b10 = x + ((b * 96 + y1) * 96 + x0) * 3;
    const float* b11 = x + ((b * 96 + y1) * 96 + x1) * 3;
#pragma unroll
    for (int c = 0; c < 3; ++c) {
      float v = w0 * b00[c] + w1 * b01[c] + w2 * b10[c] + w3 * b11[c];
      At[pp][t * 3 + c] = (_Float16)v;
    }
  }
  __syncthreads();

  // Stage 3 GEMM: pix (16x256) @ W2 (256x32), bias + ReLU -> h1
  for (int nt = 0; nt < 2; ++nt) {
    v8f acc = {0.f, 0.f, 0.f, 0.f, 0.f, 0.f, 0.f, 0.f};
#pragma unroll
    for (int s = 0; s < 8; ++s) {
      v8h lo = *(const v8h*)&At[p][s * 32 + kba];
      v8h hi = *(const v8h*)&At[p][s * 32 + 16 + kba];
      v16h a = __builtin_shufflevector(lo, hi, 0, 1, 2, 3, 4, 5, 6, 7,
                                       8, 9, 10, 11, 12, 13, 14, 15);
      v16h bf = *(const v16h*)(bpw2 + ((nt * 8 + s) * 32 + lane) * 16);
      acc = WMMA_F16(a, bf, acc);
    }
    int n = nt * 16 + nloc;
#pragma unroll
    for (int r = 0; r < 8; ++r) {
      float v = fmaxf(acc[r] + hbias[n], 0.f);
      h1[(m0 + r + hi8) * 32 + n] = (_Float16)v;
    }
  }
}

// ---------------------------------------------------------------------------
// 5x5 SAME conv 32->32 + ReLU as im2col WMMA GEMM (K=800, 25 steps x 32ch).
// One wave = TWO 16-pixel M-tiles (32 pixels) x full N=32 (2 N-tiles):
// B-fragments (shared across all waves, L2-hot) are loaded once per K-step
// and reused for both M-tiles -> 100 WMMAs per wave, B traffic halved
// relative to compute (arithmetic intensity ~16 flops/byte from L2).
// ---------------------------------------------------------------------------
__global__ __launch_bounds__(32) void conv5x5_relu_kernel(
    const _Float16* __restrict__ in, const _Float16* __restrict__ bp,
    const float* __restrict__ bias, _Float16* __restrict__ out,
    int H, int W) {
  const int lane = threadIdx.x;
  const int m0 = blockIdx.x * 32;
  const int p = lane & 15;
  // M-tile 0 pixel for this lane
  const int mA = m0 + p;
  const int jA = mA % W;
  const int iA = (mA / W) % H;
  const int bA = mA / (W * H);
  // M-tile 1 pixel for this lane
  const int mB = m0 + 16 + p;
  const int jB = mB % W;
  const int iB = (mB / W) % H;
  const int bB = mB / (W * H);
  const int kba = (lane & 16) ? 8 : 0;

  __builtin_prefetch(bp + lane * 16, 0, 3);  // global_prefetch of B pack (L2-hot)

  v8f c00 = {0.f, 0.f, 0.f, 0.f, 0.f, 0.f, 0.f, 0.f};
  v8f c01 = {0.f, 0.f, 0.f, 0.f, 0.f, 0.f, 0.f, 0.f};
  v8f c10 = {0.f, 0.f, 0.f, 0.f, 0.f, 0.f, 0.f, 0.f};
  v8f c11 = {0.f, 0.f, 0.f, 0.f, 0.f, 0.f, 0.f, 0.f};

#pragma unroll
  for (int s = 0; s < 25; ++s) {
    int dy = s / 5 - 2, dx = s % 5 - 2;

    v16h a0;
    {
      int yy = iA + dy, xx = jA + dx;
      if (yy >= 0 && yy < H && xx >= 0 && xx < W) {
        const _Float16* ap = in + (((bA * H + yy) * W) + xx) * 32;
        v8h lo = *(const v8h*)(ap + kba);
        v8h hi = *(const v8h*)(ap + 16 + kba);
        a0 = __builtin_shufflevector(lo, hi, 0, 1, 2, 3, 4, 5, 6, 7,
                                     8, 9, 10, 11, 12, 13, 14, 15);
      } else {
#pragma unroll
        for (int e = 0; e < 16; ++e) a0[e] = (_Float16)0.f;
      }
    }
    v16h a1;
    {
      int yy = iB + dy, xx = jB + dx;
      if (yy >= 0 && yy < H && xx >= 0 && xx < W) {
        const _Float16* ap = in + (((bB * H + yy) * W) + xx) * 32;
        v8h lo = *(const v8h*)(ap + kba);
        v8h hi = *(const v8h*)(ap + 16 + kba);
        a1 = __builtin_shufflevector(lo, hi, 0, 1, 2, 3, 4, 5, 6, 7,
                                     8, 9, 10, 11, 12, 13, 14, 15);
      } else {
#pragma unroll
        for (int e = 0; e < 16; ++e) a1[e] = (_Float16)0.f;
      }
    }

    v16h b0 = *(const v16h*)(bp + ((0 * 25 + s) * 32 + lane) * 16);
    v16h b1 = *(const v16h*)(bp + ((25 + s) * 32 + lane) * 16);
    c00 = WMMA_F16(a0, b0, c00);
    c01 = WMMA_F16(a0, b1, c01);
    c10 = WMMA_F16(a1, b0, c10);
    c11 = WMMA_F16(a1, b1, c11);
  }

  const int nloc = lane & 15;
  const int hi8 = (lane & 16) ? 8 : 0;
  float bn0 = bias[nloc];
  float bn1 = bias[16 + nloc];
#pragma unroll
  for (int r = 0; r < 8; ++r) {
    int mr0 = m0 + r + hi8;
    int mr1 = m0 + 16 + r + hi8;
    out[mr0 * 32 + nloc]      = (_Float16)fmaxf(c00[r] + bn0, 0.f);
    out[mr0 * 32 + 16 + nloc] = (_Float16)fmaxf(c01[r] + bn1, 0.f);
    out[mr1 * 32 + nloc]      = (_Float16)fmaxf(c10[r] + bn0, 0.f);
    out[mr1 * 32 + 16 + nloc] = (_Float16)fmaxf(c11[r] + bn1, 0.f);
  }
}

// ---------------------------------------------------------------------------
// 2x2 maxpool, f16 -> f16
// ---------------------------------------------------------------------------
__global__ void maxpool2_f16_kernel(const _Float16* __restrict__ in,
                                    _Float16* __restrict__ out, int Ho, int Wo) {
  int idx = blockIdx.x * blockDim.x + threadIdx.x;
  int total = 16 * Ho * Wo * 32;
  if (idx >= total) return;
  int c = idx & 31;
  int t = idx >> 5;
  int j = t % Wo; t /= Wo;
  int i = t % Ho;
  int b = t / Ho;
  int Wi = Wo * 2;
  const _Float16* base = in + (((b * (Ho * 2)) + 2 * i) * Wi + 2 * j) * 32 + c;
  float v0 = (float)base[0];
  float v1 = (float)base[32];
  float v2 = (float)base[Wi * 32];
  float v3 = (float)base[Wi * 32 + 32];
  out[idx] = (_Float16)fmaxf(fmaxf(v0, v1), fmaxf(v2, v3));
}

// 2x2 maxpool, f16 -> f32 (final output)
__global__ void maxpool2_out_kernel(const _Float16* __restrict__ in,
                                    float* __restrict__ out, int Ho, int Wo) {
  int idx = blockIdx.x * blockDim.x + threadIdx.x;
  int total = 16 * Ho * Wo * 32;
  if (idx >= total) return;
  int c = idx & 31;
  int t = idx >> 5;
  int j = t % Wo; t /= Wo;
  int i = t % Ho;
  int b = t / Ho;
  int Wi = Wo * 2;
  const _Float16* base = in + (((b * (Ho * 2)) + 2 * i) * Wi + 2 * j) * 32 + c;
  float v0 = (float)base[0];
  float v1 = (float)base[32];
  float v2 = (float)base[Wi * 32];
  float v3 = (float)base[Wi * 32 + 32];
  out[idx] = fmaxf(fmaxf(v0, v1), fmaxf(v2, v3));
}

// ---------------------------------------------------------------------------
extern "C" void kernel_launch(void* const* d_in, const int* in_sizes, int n_in,
                              void* d_out, int out_size, void* d_ws, size_t ws_size,
                              hipStream_t stream) {
  const float* x     = (const float*)d_in[0];   // (16,96,96,3)
  const float* fb    = (const float*)d_in[1];   // (9,9,21)
  const float* hk    = (const float*)d_in[2];   // (1,1,63,32)
  const float* hbias = (const float*)d_in[3];   // (32,)
  const float* ow    = (const float*)d_in[4];   // (9,9,3,162)
  const float* ob    = (const float*)d_in[5];   // (162,)
  const float* c2w   = (const float*)d_in[6];   // (5,5,32,32)
  const float* c2b   = (const float*)d_in[7];
  const float* c3w   = (const float*)d_in[8];
  const float* c3b   = (const float*)d_in[9];
  const float* c4w   = (const float*)d_in[10];
  const float* c4b   = (const float*)d_in[11];

  char* ws = (char*)d_ws;
  size_t o = 0;
  _Float16* bpo  = (_Float16*)(ws + o); o += (size_t)11 * 8 * 32 * 16 * 2;  // 90112
  _Float16* bpw2 = (_Float16*)(ws + o); o += (size_t)2 * 8 * 32 * 16 * 2;   // 16384
  float*    w2f  = (float*)(ws + o);    o += 32768;                          // 243*32 f32
  _Float16* bpc2 = (_Float16*)(ws + o); o += (size_t)2 * 25 * 32 * 16 * 2;  // 51200
  _Float16* bpc3 = (_Float16*)(ws + o); o += (size_t)2 * 25 * 32 * 16 * 2;
  _Float16* bpc4 = (_Float16*)(ws + o); o += (size_t)2 * 25 * 32 * 16 * 2;
  _Float16* h1   = (_Float16*)(ws + o); o += (size_t)16 * 88 * 88 * 32 * 2;  // 7.93 MB
  _Float16* h2   = (_Float16*)(ws + o); o += (size_t)16 * 88 * 88 * 32 * 2;
  _Float16* h3   = (_Float16*)(ws + o); o += (size_t)16 * 44 * 44 * 32 * 2;  // 1.98 MB
  _Float16* h4   = (_Float16*)(ws + o); o += (size_t)16 * 44 * 44 * 32 * 2;
  _Float16* h5   = (_Float16*)(ws + o); o += (size_t)16 * 44 * 44 * 32 * 2;
  // total ~22.1 MB of workspace

  // weight preparation
  compute_w2_kernel<<<(243 * 32 + 255) / 256, 256, 0, stream>>>(fb, hk, w2f);
  pack_b_kernel<<<(11 * 8 * 256) / 256, 256, 0, stream>>>(ow, bpo, 243, 162, 8, 11);
  pack_b_kernel<<<(2 * 8 * 256) / 256, 256, 0, stream>>>(w2f, bpw2, 243, 32, 8, 2);
  pack_b_kernel<<<(2 * 25 * 256) / 256, 256, 0, stream>>>(c2w, bpc2, 800, 32, 25, 2);
  pack_b_kernel<<<(2 * 25 * 256) / 256, 256, 0, stream>>>(c3w, bpc3, 800, 32, 25, 2);
  pack_b_kernel<<<(2 * 25 * 256) / 256, 256, 0, stream>>>(c4w, bpc4, 800, 32, 25, 2);

  // fused deformable layer: 16*88*88/16 = 7744 tiles
  deform_kernel<<<7744, 32, 0, stream>>>(x, bpo, bpw2, ob, hbias, h1);

  // conv2 (88x88) -> pool -> conv3 -> conv4 (44x44) -> pool(out f32)
  // 32 pixels per wave: 123904/32 = 3872 blocks, 30976/32 = 968 blocks
  conv5x5_relu_kernel<<<3872, 32, 0, stream>>>(h1, bpc2, c2b, h2, 88, 88);
  maxpool2_f16_kernel<<<(16 * 44 * 44 * 32) / 256, 256, 0, stream>>>(h2, h3, 44, 44);
  conv5x5_relu_kernel<<<968, 32, 0, stream>>>(h3, bpc3, c3b, h4, 44, 44);
  conv5x5_relu_kernel<<<968, 32, 0, stream>>>(h4, bpc4, c4b, h5, 44, 44);
  maxpool2_out_kernel<<<(16 * 22 * 22 * 32) / 256, 256, 0, stream>>>(h5, (float*)d_out, 22, 22);
}
